// PointNet_Plus_Plus_Discriminator_60971355734060
// MI455X (gfx1250) — compile-verified, matrix operand order checked by reading, not run on hardware
//
#include <hip/hip_runtime.h>

typedef _Float16 v16h __attribute__((ext_vector_type(16)));
typedef _Float16 v8h  __attribute__((ext_vector_type(8)));
typedef float    v8f  __attribute__((ext_vector_type(8)));

#define BATCH 32
#define NPTS  2048
#define S1    512
#define K1    64
#define S2    128
#define K2    64

// ---------------------------------------------------------------------------
// Blocked activation layout (fragment-native, f16):
//   element (k, col) lives at  (T*KB + kb)*512 + lane*16 + h
//   T = col/16, kb = k/32, krel = k%32,
//   khalf = (krel>>3)&1, h = ((krel&16)>>1)|(krel&7), lane = (col&15)|(khalf<<4)
// A wave reading its WMMA B-fragment reads 16 contiguous halves at lane*16.
// ---------------------------------------------------------------------------
__device__ __forceinline__ size_t blk_addr(int k, int col, int KB) {
  int kb    = k >> 5;
  int krel  = k & 31;
  int khalf = (krel >> 3) & 1;
  int h     = ((krel & 16) >> 1) | (krel & 7);
  int lane  = (col & 15) | (khalf << 4);
  return ((size_t)((col >> 4) * KB + kb) << 9) + (lane << 4) + h;
}

// ---------------------------------------------------------------------------
// Farthest point sampling: one block per batch, points staged in LDS.
// ---------------------------------------------------------------------------
__global__ void fps_kernel(const float* __restrict__ xyz, int* __restrict__ idx,
                           int Npts, int npoint) {
  __shared__ float sx[2048], sy[2048], sz[2048];
  __shared__ float rbest[256];
  __shared__ int   ribest[256];
  __shared__ int   sfar;
  const int b   = blockIdx.x;
  const int tid = threadIdx.x;
  const float* base = xyz + (size_t)b * Npts * 3;
  for (int j = tid; j < Npts; j += 256) {
    sx[j] = base[j * 3 + 0];
    sy[j] = base[j * 3 + 1];
    sz[j] = base[j * 3 + 2];
  }
  float dloc[8];
#pragma unroll
  for (int t = 0; t < 8; ++t) dloc[t] = 1e10f;
  if (tid == 0) sfar = 0;
  __syncthreads();
  for (int i = 0; i < npoint; ++i) {
    const int far = sfar;
    if (tid == 0) idx[b * npoint + i] = far;
    const float cx = sx[far], cy = sy[far], cz = sz[far];
    float best = -1.0f;
    int   besti = 0;
#pragma unroll
    for (int t = 0; t < 8; ++t) {
      int j = tid + t * 256;
      if (j < Npts) {
        float dx = sx[j] - cx, dy = sy[j] - cy, dz = sz[j] - cz;
        float dd = dx * dx + dy * dy + dz * dz;
        float dm = fminf(dloc[t], dd);
        dloc[t]  = dm;
        if (dm > best) { best = dm; besti = j; }
      }
    }
    rbest[tid]  = best;
    ribest[tid] = besti;
    __syncthreads();
    for (int s = 128; s > 0; s >>= 1) {
      if (tid < s) {
        if (rbest[tid + s] > rbest[tid]) {
          rbest[tid]  = rbest[tid + s];
          ribest[tid] = ribest[tid + s];
        }
      }
      __syncthreads();
    }
    if (tid == 0) sfar = ribest[0];
    __syncthreads();
  }
}

// out[(b*S+s)*3+c] = xyz[(b*N + idx[b*S+s])*3 + c]
__global__ void gather_xyz_kernel(const float* __restrict__ xyz,
                                  const int* __restrict__ idx,
                                  float* __restrict__ out,
                                  int N, int S, int total) {
  int i = blockIdx.x * blockDim.x + threadIdx.x;
  if (i >= total) return;
  int c = i % 3;
  int s = (i / 3) % S;
  int b = i / (3 * S);
  int j = idx[b * S + s];
  out[i] = xyz[((size_t)b * N + j) * 3 + c];
}

// Scan indices ascending, keep first nsample in ball, pad with first hit (or 0).
__global__ void ball_query_kernel(const float* __restrict__ xyz,
                                  const float* __restrict__ centers,
                                  int* __restrict__ idx,
                                  int N, int S, int nsample, float r2, int total) {
  int i = blockIdx.x * blockDim.x + threadIdx.x;   // b*S + s
  if (i >= total) return;
  int b = i / S;
  const float cx = centers[i * 3 + 0];
  const float cy = centers[i * 3 + 1];
  const float cz = centers[i * 3 + 2];
  const float* pb = xyz + (size_t)b * N * 3;
  int* ob = idx + (size_t)i * nsample;
  int cnt = 0, first = -1;
  for (int j = 0; j < N; ++j) {
    float dx = pb[j * 3 + 0] - cx;
    float dy = pb[j * 3 + 1] - cy;
    float dz = pb[j * 3 + 2] - cz;
    if (dx * dx + dy * dy + dz * dz <= r2) {
      if (cnt == 0) first = j;
      ob[cnt] = j;
      if (++cnt >= nsample) break;
    }
  }
  int fill = (first < 0) ? 0 : first;
  for (int k = cnt; k < nsample; ++k) ob[k] = fill;
}

// Vectorized zero fill (16B granules).
__global__ void zero_kernel(float4* __restrict__ p, size_t n16) {
  size_t i = (size_t)blockIdx.x * blockDim.x + threadIdx.x;
  if (i < n16) p[i] = make_float4(0.f, 0.f, 0.f, 0.f);
}

// X1 (blocked, KB=1): 3 relative-coordinate channels; rows 3..31 pre-zeroed.
__global__ void build_x1_kernel(const float* __restrict__ xyz,
                                const float* __restrict__ new_xyz,
                                const int* __restrict__ idx,
                                _Float16* __restrict__ X,
                                int N, int S, int K, int total) {
  int i = blockIdx.x * blockDim.x + threadIdx.x;  // col = (b*S+s)*K + k
  if (i >= total) return;
  int bs = i / K;
  int b  = bs / S;
  int j  = idx[i];
#pragma unroll
  for (int c = 0; c < 3; ++c) {
    float v = xyz[((size_t)b * N + j) * 3 + c] - new_xyz[(size_t)bs * 3 + c];
    X[blk_addr(c, i, 1)] = (_Float16)v;
  }
}

// X2[blk(c, dstcol)] = F[blk(c, b*Sin + idx[dstcol])]   (both blocked, KB=C/32)
__global__ void gather_feat_kernel(const _Float16* __restrict__ F,
                                   const int* __restrict__ idx,
                                   _Float16* __restrict__ X,
                                   int C, int Sin, int S, int K, int total) {
  int i = blockIdx.x * blockDim.x + threadIdx.x;  // dst col
  if (i >= total) return;
  int bs = i / K;
  int b  = bs / S;
  int j  = idx[i];
  int src = b * Sin + j;
  int KB = C >> 5;
  for (int c = 0; c < C; ++c)
    X[blk_addr(c, i, KB)] = F[blk_addr(c, src, KB)];
}

// Fold BN into weights: Wh[co][ciPad] = W[co][ci]*gamma/sqrt(var+eps), zero pad;
// bias = beta - mean*scale. FC layers (gamma==null): scale=1, bias=b.
__global__ void prep_weights_kernel(const float* __restrict__ W,
                                    const float* __restrict__ gamma,
                                    const float* __restrict__ beta,
                                    const float* __restrict__ mean,
                                    const float* __restrict__ var,
                                    const float* __restrict__ bvec,
                                    _Float16* __restrict__ Wh,
                                    float* __restrict__ bias,
                                    int Cin, int CinPad, int Cout) {
  int i = blockIdx.x * blockDim.x + threadIdx.x;
  if (i >= Cout * CinPad) return;
  int co = i / CinPad;
  int ci = i % CinPad;
  float scale, bo;
  if (gamma) {
    scale = gamma[co] * rsqrtf(var[co] + 1e-5f);
    bo    = beta[co] - mean[co] * scale;
  } else {
    scale = 1.0f;
    bo    = bvec[co];
  }
  float w = (ci < Cin) ? W[(size_t)co * Cin + ci] * scale : 0.0f;
  Wh[(size_t)co * CinPad + ci] = (_Float16)w;
  if (ci == 0) bias[co] = bo;
}

// ---------------------------------------------------------------------------
// Fused GEMM + bias + ReLU on v_wmma_f32_16x16x32_f16, blocked layouts.
// Wave = 16(M) x (NTT*16)(N) tiles sharing one A fragment per k-step.
// Launch guarantees every covered tile is full -> branch-free inner loop.
// Software pipelined: kb+1 fragments are in flight while kb's WMMAs run.
// ---------------------------------------------------------------------------
template <int NTT>
__global__ void gemm_wmma_blk(const _Float16* __restrict__ Wh,
                              const float* __restrict__ bias,
                              const _Float16* __restrict__ X,
                              _Float16* __restrict__ Y,
                              int KB, int KBout, int Ncols, int doRelu) {
  const int lane = threadIdx.x & 31;
  const int wave = threadIdx.x >> 5;
  const int m0   = blockIdx.y * 16;
  const int T0   = (blockIdx.x * 4 + wave) * NTT;
  if (T0 * 16 >= Ncols) return;
  const int khalf  = lane >> 4;
  const int nl     = lane & 15;
  const int CinPad = KB << 5;

  const _Float16* wrow  = Wh + (size_t)(m0 + nl) * CinPad;
  const _Float16* xbase = X + (((size_t)T0 * KB) << 9) + (lane << 4);

  v8f acc[NTT];
#pragma unroll
  for (int t = 0; t < NTT; ++t) acc[t] = (v8f){};

  v16h aCur, aNxt = {};
  v16h bCur[NTT], bNxt[NTT];
#pragma unroll
  for (int t = 0; t < NTT; ++t) bNxt[t] = (v16h){};

  // prologue: fragments for kb = 0
  {
    const v8h* wp = (const v8h*)(wrow + khalf * 8);
    aCur = __builtin_shufflevector(wp[0], wp[2], 0, 1, 2, 3, 4, 5, 6, 7,
                                   8, 9, 10, 11, 12, 13, 14, 15);
#pragma unroll
    for (int t = 0; t < NTT; ++t) {
      const v8h* bp = (const v8h*)(xbase + (((size_t)t * KB) << 9));
      bCur[t] = __builtin_shufflevector(bp[0], bp[1], 0, 1, 2, 3, 4, 5, 6, 7,
                                        8, 9, 10, 11, 12, 13, 14, 15);
    }
  }

  for (int kb = 0; kb < KB; ++kb) {
    const int kn = (kb + 1 < KB) ? kb + 1 : kb;   // branchless clamp
    {
      const v8h* wp = (const v8h*)(wrow + (kn << 5) + khalf * 8);
      aNxt = __builtin_shufflevector(wp[0], wp[2], 0, 1, 2, 3, 4, 5, 6, 7,
                                     8, 9, 10, 11, 12, 13, 14, 15);
#pragma unroll
      for (int t = 0; t < NTT; ++t) {
        const v8h* bp = (const v8h*)(xbase + (((size_t)t * KB + kn) << 9));
        bNxt[t] = __builtin_shufflevector(bp[0], bp[1], 0, 1, 2, 3, 4, 5, 6, 7,
                                          8, 9, 10, 11, 12, 13, 14, 15);
      }
      __builtin_prefetch(xbase + (((size_t)kn + 1) << 9), 0, 1);
    }
#pragma unroll
    for (int t = 0; t < NTT; ++t)
      // (neg_a, A, neg_b, B, c_mod, C, reuse_a, reuse_b)
      acc[t] = __builtin_amdgcn_wmma_f32_16x16x32_f16(false, aCur, false,
                                                      bCur[t], (short)0,
                                                      acc[t], false, false);
    aCur = aNxt;
#pragma unroll
    for (int t = 0; t < NTT; ++t) bCur[t] = bNxt[t];
  }

#pragma unroll
  for (int i = 0; i < 8; ++i) {
    const int   m  = m0 + khalf * 8 + i;
    const float bv = bias[m];
#pragma unroll
    for (int t = 0; t < NTT; ++t) {
      float v = acc[t][i] + bv;
      if (doRelu) v = fmaxf(v, 0.0f);
      Y[blk_addr(m, (T0 + t) * 16 + nl, KBout)] = (_Float16)v;
    }
  }
}

// Blocked maxpool over Kg contiguous columns: Y(c,g) = max_k X(c, g*Kg+k).
__global__ void maxpool_blk_kernel(const _Float16* __restrict__ X,
                                   _Float16* __restrict__ Y,
                                   int C, int G, int Kg) {
  int i = blockIdx.x * blockDim.x + threadIdx.x;
  if (i >= C * G) return;
  int c  = i / G;
  int g  = i % G;
  int KB = C >> 5;
  float m = -1e30f;
  for (int k = 0; k < Kg; ++k)
    m = fmaxf(m, (float)X[blk_addr(c, g * Kg + k, KB)]);
  Y[blk_addr(c, g, KB)] = (_Float16)m;
}

// Final FC (Cout=1): out[b] = W . H(:,b) + bias, H blocked with KB=C/32.
__global__ void fc3_kernel(const float* __restrict__ W,
                           const float* __restrict__ bvec,
                           const _Float16* __restrict__ H,
                           float* __restrict__ out, int Cin, int B) {
  int b = threadIdx.x;
  if (b >= B) return;
  int KB = Cin >> 5;
  float s = bvec[0];
  for (int c = 0; c < Cin; ++c)
    s += W[c] * (float)H[blk_addr(c, b, KB)];
  out[b] = s;
}

// ---------------------------------------------------------------------------
extern "C" void kernel_launch(void* const* d_in, const int* in_sizes, int n_in,
                              void* d_out, int out_size, void* d_ws, size_t ws_size,
                              hipStream_t stream) {
  (void)in_sizes; (void)n_in; (void)out_size; (void)ws_size;

  const float* x = (const float*)d_in[0];

  char*  ws  = (char*)d_ws;
  size_t off = 0;
  auto alloc = [&](size_t bytes) -> void* {
    void* p = ws + off;
    off = (off + bytes + 255) & ~(size_t)255;
    return p;
  };

  int*      fps1  = (int*)     alloc((size_t)BATCH * S1 * 4);
  float*    nxyz  = (float*)   alloc((size_t)BATCH * S1 * 3 * 4);
  int*      bq1   = (int*)     alloc((size_t)BATCH * S1 * K1 * 4);
  int*      fps2  = (int*)     alloc((size_t)BATCH * S2 * 4);
  float*    xyz2  = (float*)   alloc((size_t)BATCH * S2 * 3 * 4);
  int*      bq2   = (int*)     alloc((size_t)BATCH * S2 * K2 * 4);
  _Float16* fsa1  = (_Float16*)alloc((size_t)128 * BATCH * S1 * 2);
  _Float16* f2b   = (_Float16*)alloc((size_t)256 * BATCH * S2 * 2);
  _Float16* hbuf  = (_Float16*)alloc((size_t)1024 * BATCH * 2);
  _Float16* h1    = (_Float16*)alloc((size_t)512 * BATCH * 2);
  _Float16* h2    = (_Float16*)alloc((size_t)256 * BATCH * 2);

  // 11 WMMA layers: {d_in base, has_bn, Cin, CinPad, Cout}
  struct Lyr { int base; int bn; int cin, cinp, cout; _Float16* Wh; float* bias; };
  Lyr L[11] = {
    { 1, 1,    3,   32,   64, nullptr, nullptr},   // sa1.0
    { 6, 1,   64,   64,   64, nullptr, nullptr},   // sa1.1
    {11, 1,   64,   64,  128, nullptr, nullptr},   // sa1.2
    {16, 1,  128,  128,  128, nullptr, nullptr},   // sa2.0
    {21, 1,  128,  128,  128, nullptr, nullptr},   // sa2.1
    {26, 1,  128,  128,  256, nullptr, nullptr},   // sa2.2
    {31, 1,  256,  256,  256, nullptr, nullptr},   // sa3.0
    {36, 1,  256,  256,  512, nullptr, nullptr},   // sa3.1
    {41, 1,  512,  512, 1024, nullptr, nullptr},   // sa3.2
    {46, 0, 1024, 1024,  512, nullptr, nullptr},   // fc1
    {48, 0,  512,  512,  256, nullptr, nullptr},   // fc2
  };
  for (int i = 0; i < 11; ++i) {
    L[i].Wh   = (_Float16*)alloc((size_t)L[i].cout * L[i].cinp * 2);
    L[i].bias = (float*)   alloc((size_t)L[i].cout * 4);
  }

  _Float16* R1 = (_Float16*)alloc((size_t)256 << 20);  // ping
  _Float16* R2 = (_Float16*)alloc((size_t)256 << 20);  // pong

  // ---- weight prep ----------------------------------------------------
  for (int i = 0; i < 11; ++i) {
    int total = L[i].cout * L[i].cinp;
    const float* Wp = (const float*)d_in[L[i].base];
    const float* g  = L[i].bn ? (const float*)d_in[L[i].base + 1] : nullptr;
    const float* be = L[i].bn ? (const float*)d_in[L[i].base + 2] : nullptr;
    const float* me = L[i].bn ? (const float*)d_in[L[i].base + 3] : nullptr;
    const float* va = L[i].bn ? (const float*)d_in[L[i].base + 4] : nullptr;
    const float* bv = L[i].bn ? nullptr : (const float*)d_in[L[i].base + 1];
    prep_weights_kernel<<<(total + 255) / 256, 256, 0, stream>>>(
        Wp, g, be, me, va, bv, L[i].Wh, L[i].bias, L[i].cin, L[i].cinp, L[i].cout);
  }

  auto gemm = [&](int li, const _Float16* X, _Float16* Y, int Ncols, int relu) {
    int nTiles = Ncols / 16;
    if (nTiles % 16 == 0) {          // all conv layers: exact NT=4 coverage
      dim3 grid((unsigned)(nTiles / 16), (unsigned)(L[li].cout / 16));
      gemm_wmma_blk<4><<<grid, 128, 0, stream>>>(L[li].Wh, L[li].bias, X, Y,
                                                 L[li].cinp / 32, L[li].cout / 32,
                                                 Ncols, relu);
    } else {                         // FC head (Ncols == 32): NT=2
      dim3 grid(1u, (unsigned)(L[li].cout / 16));
      gemm_wmma_blk<2><<<grid, 128, 0, stream>>>(L[li].Wh, L[li].bias, X, Y,
                                                 L[li].cinp / 32, L[li].cout / 32,
                                                 Ncols, relu);
    }
  };

  // ---- SA1 ------------------------------------------------------------
  fps_kernel<<<BATCH, 256, 0, stream>>>(x, fps1, NPTS, S1);
  {
    int t = BATCH * S1 * 3;
    gather_xyz_kernel<<<(t + 255) / 256, 256, 0, stream>>>(x, fps1, nxyz, NPTS, S1, t);
  }
  {
    int t = BATCH * S1;
    ball_query_kernel<<<(t + 255) / 256, 256, 0, stream>>>(x, nxyz, bq1, NPTS, S1,
                                                           K1, 0.04f, t);
  }
  const int N1 = BATCH * S1 * K1;   // 1,048,576 columns
  {
    size_t n16 = ((size_t)32 * N1 * 2) / 16;   // zero padded X1 (KB=1)
    zero_kernel<<<(unsigned)((n16 + 255) / 256), 256, 0, stream>>>((float4*)R1, n16);
  }
  build_x1_kernel<<<(N1 + 255) / 256, 256, 0, stream>>>(x, nxyz, bq1, R1,
                                                        NPTS, S1, K1, N1);
  gemm(0, R1, R2, N1, 1);
  gemm(1, R2, R1, N1, 1);
  gemm(2, R1, R2, N1, 1);
  {
    int t = 128 * BATCH * S1;
    maxpool_blk_kernel<<<(t + 255) / 256, 256, 0, stream>>>(R2, fsa1, 128,
                                                            BATCH * S1, K1);
  }

  // ---- SA2 ------------------------------------------------------------
  fps_kernel<<<BATCH, 256, 0, stream>>>(nxyz, fps2, S1, S2);
  {
    int t = BATCH * S2 * 3;
    gather_xyz_kernel<<<(t + 255) / 256, 256, 0, stream>>>(nxyz, fps2, xyz2, S1, S2, t);
  }
  {
    int t = BATCH * S2;
    ball_query_kernel<<<(t + 255) / 256, 256, 0, stream>>>(nxyz, xyz2, bq2, S1, S2,
                                                           K2, 0.16f, t);
  }
  const int N2 = BATCH * S2 * K2;   // 262,144 columns
  gather_feat_kernel<<<(N2 + 255) / 256, 256, 0, stream>>>(fsa1, bq2, R1, 128,
                                                           S1, S2, K2, N2);
  gemm(3, R1, R2, N2, 1);
  gemm(4, R2, R1, N2, 1);
  gemm(5, R1, R2, N2, 1);
  {
    int t = 256 * BATCH * S2;
    maxpool_blk_kernel<<<(t + 255) / 256, 256, 0, stream>>>(R2, f2b, 256,
                                                            BATCH * S2, K2);
  }

  // ---- SA3 (4096 columns) ----------------------------------------------
  const int N3 = BATCH * S2;
  gemm(6, f2b, R1, N3, 1);
  gemm(7, R1, R2, N3, 1);
  gemm(8, R2, R1, N3, 1);
  {
    int t = 1024 * BATCH;
    maxpool_blk_kernel<<<(t + 255) / 256, 256, 0, stream>>>(R1, hbuf, 1024,
                                                            BATCH, S2);
  }

  // ---- FC head ----------------------------------------------------------
  gemm(9,  hbuf, h1, BATCH, 1);
  gemm(10, h1,   h2, BATCH, 1);
  fc3_kernel<<<1, 32, 0, stream>>>((const float*)d_in[50], (const float*)d_in[51],
                                   h2, (float*)d_out, 256, BATCH);
}
